// GAT_GCN_Transformer_meth_ge_mut_multiheadattn_57836029608454
// MI455X (gfx1250) — matrix operand order repaired and checked by reading, not compiled, not run on hardware
//
#include <hip/hip_runtime.h>
#include <cmath>

// ---------------------------------------------------------------------------
// MI455X / gfx1250 implementation.
// All dense GEMMs -> v_wmma_f32_16x16x32_f16 (f16 in, f32 accumulate).
// GEMM tiles staged into LDS via the Tensor Data Mover (tensor_load_to_lds,
// TENSORcnt), weights staged TRANSPOSED so every WMMA B-fragment is two 16B
// vector loads. 8000-token self-attention -> fused flash-style kernel (online
// softmax), O-accumulator in LDS, K/V streamed from L2 (13MB << 192MB L2),
// V transposed. Graph scatter ops -> float atomics (order-preserving keys).
// ---------------------------------------------------------------------------

typedef __attribute__((ext_vector_type(16))) _Float16 v16h;
typedef __attribute__((ext_vector_type(8)))  float    v8f;
typedef __attribute__((ext_vector_type(8)))  _Float16 h8;
typedef __attribute__((ext_vector_type(4)))  unsigned u32x4;
typedef __attribute__((ext_vector_type(8)))  int      i32x8;
typedef __attribute__((ext_vector_type(4)))  int      i32x4;

union AF { v16h v; _Float16 h[16]; };
union CF { v8f  v; float    f[8];  };

#define NN   8000
#define MP   8064      /* pad128(8000) */
#define D1   78
#define D2   780
#define FFH  2048
#define NEDGE 40000    /* 32000 edges + 8000 self loops */

// A-matrix 16x32 f16 fragment (ISA 7.12.2): lane holds row m=l&15;
// element e -> k = e + 8*half (e<8), k = 16 + (e-8) + 8*half (e>=8).
__device__ __forceinline__ v16h loadA32(const _Float16* p, int half) {
  AF a;
  *(h8*)&a.h[0] = *(const h8*)(p + 8 * half);
  *(h8*)&a.h[8] = *(const h8*)(p + 16 + 8 * half);
  return a.v;
}
// B-matrix 32x16 f16 fragment: lane holds col n=l&15;
// element e -> k = e + 8*half (e<8), k = 16 + (e-8) + 8*half (e>=8).
__device__ __forceinline__ v16h loadB32(const _Float16* p, int half) {
  AF b;
  *(h8*)&b.h[0] = *(const h8*)(p + 8 * half);
  *(h8*)&b.h[8] = *(const h8*)(p + 16 + 8 * half);
  return b.v;
}

// Tensor Data Mover: DMA a 2D tile (tile_d1 lines of tile_d0 f16 elements,
// line stride = stride_el elements) from global memory into LDS at lds_addr.
// D# layout per CDNA5 ISA. Issued per-wave (EXEC
// ignored); caller guards so only one wave issues; completion = TENSORcnt.
// Toolchain: 6-arg builtin (amdgpu-toolchain clang-23 form).
__device__ __forceinline__ void tdm_load_2d(unsigned lds_addr, const void* gaddr,
                                            unsigned stride_el,
                                            unsigned tile_d0, unsigned tile_d1) {
  unsigned long long ga = (unsigned long long)(size_t)gaddr;
  const unsigned td0 = 0x7FFFFFFFu, td1 = 0x7FFFFFFFu;  // no OOB clip (padded)
  u32x4 g0;
  g0[0] = 1u;                                        // count=1, user D#
  g0[1] = lds_addr;                                  // LDS byte address
  g0[2] = (unsigned)(ga & 0xFFFFFFFFu);              // global_addr[31:0]
  g0[3] = (unsigned)((ga >> 32) & 0x01FFFFFFu)       // global_addr[56:32]
        | 0x80000000u;                               // type=2 ("image")
  i32x8 g1;
  g1[0] = 0x00010000;                                // data_size=1 (2 bytes)
  g1[1] = (int)((td0 & 0xFFFFu) << 16);              // tensor_dim0 lo
  g1[2] = (int)((td0 >> 16) | ((td1 & 0xFFFFu) << 16));
  g1[3] = (int)((td1 >> 16) | (tile_d0 << 16));      // tile_dim0
  g1[4] = (int)(tile_d1 & 0xFFFFu);                  // tile_dim1 (tile_dim2=0)
  g1[5] = (int)stride_el;                            // tensor_dim0_stride lo
  g1[6] = 0;                                         // stride hi, dim1_stride lo
  g1[7] = 0;
  i32x4 g2; g2[0] = 0; g2[1] = 0; g2[2] = 0; g2[3] = 0;
  i32x4 g3; g3[0] = 0; g3[1] = 0; g3[2] = 0; g3[3] = 0;
  i32x8 g4;
  g4[0] = 0; g4[1] = 0; g4[2] = 0; g4[3] = 0;
  g4[4] = 0; g4[5] = 0; g4[6] = 0; g4[7] = 0;
  __builtin_amdgcn_tensor_load_to_lds(g0, g1, g2, g3, g4, 0);
}

__device__ __forceinline__ float rmax16(float x) {
  for (int m = 1; m < 16; m <<= 1) x = fmaxf(x, __shfl_xor(x, m, 32));
  return x;
}
__device__ __forceinline__ float rsum16(float x) {
  for (int m = 1; m < 16; m <<= 1) x += __shfl_xor(x, m, 32);
  return x;
}
__device__ __forceinline__ unsigned fkey(float f) {   // order-preserving map
  unsigned b = __float_as_uint(f);
  return (b & 0x80000000u) ? ~b : (b | 0x80000000u);
}
__device__ __forceinline__ float funkey(unsigned u) {
  unsigned b = (u & 0x80000000u) ? (u ^ 0x80000000u) : ~u;
  return __uint_as_float(b);
}
__device__ __forceinline__ void edge_sd(const int* ei, int e, int& s, int& d) {
  if (e < 32000) { s = ei[e]; d = ei[32000 + e]; }
  else           { s = e - 32000; d = e - 32000; }
}

// --------------------------- fp32 -> f16 pad-copy ---------------------------
__global__ void k_cvt_pad(const float* __restrict__ s, _Float16* __restrict__ d,
                          int R, int C, int Rp, int Cp) {
  int i = blockIdx.x * blockDim.x + threadIdx.x;
  int total = Rp * Cp;
  if (i >= total) return;
  int r = i / Cp, c = i % Cp;
  d[i] = (r < R && c < C) ? (_Float16)s[(size_t)r * C + c] : (_Float16)0.0f;
}
// transposed: d is Cp x Rp, d[c*Rp + r] = s[r*C + c]
__global__ void k_cvt_padT(const float* __restrict__ s, _Float16* __restrict__ d,
                           int R, int C, int Rp, int Cp) {
  int i = blockIdx.x * blockDim.x + threadIdx.x;
  int total = Cp * Rp;
  if (i >= total) return;
  int c = i / Rp, r = i % Rp;
  d[i] = (r < R && c < C) ? (_Float16)s[(size_t)r * C + c] : (_Float16)0.0f;
}

// ------------------------------- WMMA GEMM ---------------------------------
// D[M,N] = A[M,Kp] @ B[Kp,N] (+bias)(+relu).
// A: f16 row-major Mp x lda.  BT: f16 *transposed* Np x ldbt (BT[n][k]).
// Block 256 thr = 8 waves, tile 128x64; wave -> 32x32 (2x2 WMMA 16x16x32).
// Tile staging global->LDS done by the TDM (wave 0 issues descriptors).
__global__ __launch_bounds__(256) void k_gemm(
    const _Float16* __restrict__ A, const _Float16* __restrict__ BT,
    float* __restrict__ D, const float* __restrict__ bias,
    int M, int N, int Kp, int lda, int ldbt, int relu) {
  __shared__ _Float16 sA[128 * 32];   // [row][k]
  __shared__ _Float16 sBt[64 * 32];   // [col][k]
  const int t = threadIdx.x;
  const int lane = t & 31, wid = t >> 5;
  const int wm = wid & 3, wn = wid >> 2;
  const int half = lane >> 4, ln = lane & 15;
  const int blockM = blockIdx.y * 128;
  const int blockN = blockIdx.x * 64;

  CF acc[2][2];
  for (int i = 0; i < 2; i++)
    for (int j = 0; j < 2; j++)
      for (int r = 0; r < 8; r++) acc[i][j].f[r] = 0.0f;

  for (int k0 = 0; k0 < Kp; k0 += 32) {
    __syncthreads();                   // everyone done reading previous tiles
    if (wid == 0) {                    // TDM: DMA both tiles into LDS
      tdm_load_2d((unsigned)(size_t)sA,
                  &A[(size_t)blockM * lda + k0], (unsigned)lda, 32u, 128u);
      tdm_load_2d((unsigned)(size_t)sBt,
                  &BT[(size_t)blockN * ldbt + k0], (unsigned)ldbt, 32u, 64u);
      __builtin_amdgcn_s_wait_tensorcnt(0);
    }
    if (k0 + 32 < Kp) {                // prefetch next K-slab into caches
      __builtin_prefetch(&A[(size_t)(blockM + (t >> 1)) * lda + k0 + 32], 0, 1);
      if (t < 128)
        __builtin_prefetch(&BT[(size_t)(blockN + (t >> 1)) * ldbt + k0 + 32], 0, 1);
    }
    __syncthreads();                   // publish tiles to all waves
    AF a[2], b[2];
    for (int mi = 0; mi < 2; mi++)
      a[mi].v = loadA32(&sA[(wm * 32 + mi * 16 + ln) * 32], half);
    for (int ni = 0; ni < 2; ni++)
      b[ni].v = loadB32(&sBt[(wn * 32 + ni * 16 + ln) * 32], half);
    for (int mi = 0; mi < 2; mi++)
      for (int ni = 0; ni < 2; ni++)
        acc[mi][ni].v = __builtin_amdgcn_wmma_f32_16x16x32_f16(
            false, a[mi].v, false, b[ni].v, (short)0, acc[mi][ni].v, false, false);
  }
  for (int mi = 0; mi < 2; mi++)
    for (int ni = 0; ni < 2; ni++) {
      int r0 = blockM + wm * 32 + mi * 16 + 8 * half;
      int c0 = blockN + wn * 32 + ni * 16 + ln;
      if (c0 >= N) continue;
      #pragma unroll
      for (int r = 0; r < 8; r++) {
        int row = r0 + r;
        if (row >= M) continue;
        float v = acc[mi][ni].f[r];
        if (bias) v += bias[c0];
        if (relu) v = fmaxf(v, 0.0f);
        D[(size_t)row * N + c0] = v;
      }
    }
}

// ------------------------- fused flash attention ---------------------------
// O = softmax(Q K^T * scale) V.  1 wave / 16 query rows. dpad mult of 32.
// Q,K: row-major [rows x dpad].  VT: transposed [dpad x nkp] (feature-major).
__global__ __launch_bounds__(32) void k_attn(
    const _Float16* __restrict__ Q, const _Float16* __restrict__ K,
    const _Float16* __restrict__ VT, float* __restrict__ O,
    int nq, int nk, int nkp, int dtrue, int dpad, float scale) {
  extern __shared__ char smem[];
  float*    Oacc = (float*)smem;                                  // 16*dpad f32
  _Float16* Pst  = (_Float16*)(smem + (size_t)16 * dpad * 4);     // 16x32 f16
  float*    ls   = (float*)(smem + (size_t)16 * dpad * 4 + 16 * 32 * 2);
  const int lane = threadIdx.x;
  const int half = lane >> 4, ln = lane & 15;
  const int qbase = blockIdx.x * 16;

  for (int i = lane; i < 16 * dpad; i += 32) Oacc[i] = 0.0f;
  __syncthreads();

  float mrow[8], lrow[8];
  #pragma unroll
  for (int r = 0; r < 8; r++) { mrow[r] = -1e30f; lrow[r] = 0.0f; }

  for (int j = 0; j < nk; j += 32) {
    CF s0, s1;
    for (int r = 0; r < 8; r++) { s0.f[r] = 0.0f; s1.f[r] = 0.0f; }
    for (int k0 = 0; k0 < dpad; k0 += 32) {        // Q K^T over features
      AF a, b0, b1;
      a.v  = loadA32(&Q[(size_t)(qbase + ln) * dpad + k0], half);
      b0.v = loadB32(&K[(size_t)(j + ln) * dpad + k0], half);
      b1.v = loadB32(&K[(size_t)(j + 16 + ln) * dpad + k0], half);
      s0.v = __builtin_amdgcn_wmma_f32_16x16x32_f16(false, a.v, false, b0.v,
                                                    (short)0, s0.v, false, false);
      s1.v = __builtin_amdgcn_wmma_f32_16x16x32_f16(false, a.v, false, b1.v,
                                                    (short)0, s1.v, false, false);
    }
    float corr[8];
    #pragma unroll
    for (int r = 0; r < 8; r++) {                  // online softmax update
      float v0 = s0.f[r] * scale, v1 = s1.f[r] * scale;
      float tm = rmax16(fmaxf(v0, v1));
      float nm = fmaxf(mrow[r], tm);
      float e0 = expf(v0 - nm), e1 = expf(v1 - nm);
      float rs = rsum16(e0 + e1);
      corr[r] = expf(mrow[r] - nm);
      lrow[r] = lrow[r] * corr[r] + rs;
      mrow[r] = nm;
      int mr = r + 8 * half;
      Pst[mr * 32 + ln]      = (_Float16)e0;
      Pst[mr * 32 + 16 + ln] = (_Float16)e1;
    }
    __syncthreads();
    AF pa;
    pa.v = loadA32(&Pst[ln * 32], half);
    for (int fb = 0; fb < dpad; fb += 16) {        // P @ V, corr folded in
      AF bv;
      bv.v = loadB32(&VT[(size_t)(fb + ln) * nkp + j], half);
      CF c;
      #pragma unroll
      for (int r = 0; r < 8; r++)
        c.f[r] = Oacc[(r + 8 * half) * dpad + fb + ln] * corr[r];
      c.v = __builtin_amdgcn_wmma_f32_16x16x32_f16(false, pa.v, false, bv.v,
                                                   (short)0, c.v, false, false);
      #pragma unroll
      for (int r = 0; r < 8; r++)
        Oacc[(r + 8 * half) * dpad + fb + ln] = c.f[r];
    }
    __syncthreads();
  }
  #pragma unroll
  for (int r = 0; r < 8; r++) ls[r + 8 * half] = lrow[r];
  __syncthreads();
  for (int i = lane; i < 16 * dtrue; i += 32) {
    int row = i / dtrue, col = i % dtrue;
    if (qbase + row < nq)
      O[(size_t)(qbase + row) * dtrue + col] = Oacc[row * dpad + col] / ls[row];
  }
}

// ------------------------- residual + layernorm ----------------------------
__global__ __launch_bounds__(256) void k_resln(
    const float* __restrict__ x, const float* __restrict__ y,
    const float* __restrict__ g, const float* __restrict__ b,
    float* __restrict__ o, int d) {
  __shared__ float s1[256], s2[256], mv[2];
  int row = blockIdx.x;
  size_t base = (size_t)row * d;
  float a = 0.0f, q = 0.0f;
  for (int c = threadIdx.x; c < d; c += 256) {
    float v = x[base + c] + y[base + c];
    a += v; q += v * v;
  }
  s1[threadIdx.x] = a; s2[threadIdx.x] = q;
  __syncthreads();
  for (int s = 128; s > 0; s >>= 1) {
    if (threadIdx.x < s) { s1[threadIdx.x] += s1[threadIdx.x + s];
                           s2[threadIdx.x] += s2[threadIdx.x + s]; }
    __syncthreads();
  }
  if (threadIdx.x == 0) {
    float mean = s1[0] / d;
    float var  = s2[0] / d - mean * mean;
    mv[0] = mean; mv[1] = rsqrtf(fmaxf(var, 0.0f) + 1e-5f);
  }
  __syncthreads();
  float mean = mv[0], inv = mv[1];
  for (int c = threadIdx.x; c < d; c += 256) {
    float v = x[base + c] + y[base + c];
    o[base + c] = (v - mean) * inv * g[c] + b[c];
  }
}

// ------------------------------ small kernels ------------------------------
__global__ void k_fillf(float* p, float v, int n) {
  int i = blockIdx.x * blockDim.x + threadIdx.x; if (i < n) p[i] = v;
}
__global__ void k_fillu(unsigned* p, unsigned v, int n) {
  int i = blockIdx.x * blockDim.x + threadIdx.x; if (i < n) p[i] = v;
}
__global__ void k_biasrelu(float* __restrict__ h, const float* __restrict__ b,
                           int M, int d) {
  int i = blockIdx.x * blockDim.x + threadIdx.x;
  if (i < M * d) { float v = h[i] + b[i % d]; h[i] = fmaxf(v, 0.0f); }
}

// -------------------------------- GAT -------------------------------------
__global__ void k_gatcoef(const float* __restrict__ hw,
                          const float* __restrict__ asrc,
                          const float* __restrict__ adst,
                          float* __restrict__ AS, float* __restrict__ AD, int N) {
  int i = blockIdx.x * blockDim.x + threadIdx.x;
  if (i >= N * 10) return;
  int n = i / 10, hd = i % 10;
  const float* hp = hw + (size_t)n * 780 + hd * 78;
  const float* sp = asrc + hd * 78;
  const float* dp = adst + hd * 78;
  float as = 0.0f, ad = 0.0f;
  for (int f = 0; f < 78; f++) { as += hp[f] * sp[f]; ad += hp[f] * dp[f]; }
  AS[i] = as; AD[i] = ad;
}
__global__ void k_gatmax(const int* __restrict__ ei, const float* __restrict__ AS,
                         const float* __restrict__ AD, unsigned* __restrict__ mk) {
  int i = blockIdx.x * blockDim.x + threadIdx.x;
  if (i >= NEDGE * 10) return;
  int e = i / 10, hd = i % 10, s, d;
  edge_sd(ei, e, s, d);
  float v = AS[s * 10 + hd] + AD[d * 10 + hd];
  v = (v >= 0.0f) ? v : 0.2f * v;
  atomicMax(&mk[d * 10 + hd], fkey(v));
}
__global__ void k_gatexp(const int* __restrict__ ei, const float* __restrict__ AS,
                         const float* __restrict__ AD, const unsigned* __restrict__ mk,
                         float* __restrict__ exs, float* __restrict__ den) {
  int i = blockIdx.x * blockDim.x + threadIdx.x;
  if (i >= NEDGE * 10) return;
  int e = i / 10, hd = i % 10, s, d;
  edge_sd(ei, e, s, d);
  float v = AS[s * 10 + hd] + AD[d * 10 + hd];
  v = (v >= 0.0f) ? v : 0.2f * v;
  float ex = expf(v - funkey(mk[d * 10 + hd]));
  exs[i] = ex;
  atomicAdd(&den[d * 10 + hd], ex);
}
__global__ void k_gatscat(const int* __restrict__ ei, const float* __restrict__ exs,
                          const float* __restrict__ den, const float* __restrict__ hw,
                          float* __restrict__ out) {
  int i = blockIdx.x * blockDim.x + threadIdx.x;
  if (i >= NEDGE * 780) return;
  int e = i / 780, c = i % 780, hd = c / 78, s, d;
  edge_sd(ei, e, s, d);
  float alpha = exs[e * 10 + hd] / den[d * 10 + hd];
  atomicAdd(&out[(size_t)d * 780 + c], alpha * hw[(size_t)s * 780 + c]);
}

// -------------------------------- GCN -------------------------------------
__global__ void k_deg(const int* __restrict__ ei, float* __restrict__ deg) {
  int e = blockIdx.x * blockDim.x + threadIdx.x;
  if (e >= NEDGE) return;
  int s, d; edge_sd(ei, e, s, d);
  atomicAdd(&deg[d], 1.0f);
}
__global__ void k_gcnscat(const int* __restrict__ ei, const float* __restrict__ deg,
                          const float* __restrict__ hw, float* __restrict__ out) {
  int i = blockIdx.x * blockDim.x + threadIdx.x;
  if (i >= NEDGE * 780) return;
  int e = i / 780, c = i % 780, s, d;
  edge_sd(ei, e, s, d);
  float norm = rsqrtf(deg[s]) * rsqrtf(deg[d]);
  atomicAdd(&out[(size_t)d * 780 + c], norm * hw[(size_t)s * 780 + c]);
}

// ------------------------------- pooling -----------------------------------
__global__ void k_cnt(const int* __restrict__ batch, float* __restrict__ cnt, int N) {
  int n = blockIdx.x * blockDim.x + threadIdx.x;
  if (n < N) atomicAdd(&cnt[batch[n]], 1.0f);
}
__global__ void k_pool(const float* __restrict__ h, const int* __restrict__ batch,
                       unsigned* __restrict__ gmax, float* __restrict__ gsum,
                       int N, int d) {
  int i = blockIdx.x * blockDim.x + threadIdx.x;
  if (i >= N * d) return;
  int n = i / d, c = i % d;
  int b = batch[n];
  float v = h[i];
  atomicMax(&gmax[b * d + c], fkey(v));
  atomicAdd(&gsum[b * d + c], v);
}
__global__ void k_xg(const unsigned* __restrict__ gmax, const float* __restrict__ gsum,
                     const float* __restrict__ cnt, float* __restrict__ xg, int d) {
  int i = blockIdx.x * blockDim.x + threadIdx.x;
  if (i >= 256 * 2 * d) return;
  int b = i / (2 * d), c = i % (2 * d);
  xg[i] = (c < d) ? funkey(gmax[b * d + c]) : gsum[b * d + (c - d)] / cnt[b];
}

// ------------------------------ conv branch --------------------------------
__global__ void k_conv1d(const float* __restrict__ in, const float* __restrict__ w,
                         const float* __restrict__ bias, float* __restrict__ out,
                         int Bn, int Cin, int Lin, int Cout, int Lout) {
  int i = blockIdx.x * blockDim.x + threadIdx.x;
  if (i >= Bn * Cout * Lout) return;
  int p = i % Lout, co = (i / Lout) % Cout, b = i / (Lout * Cout);
  float acc = bias[co];
  for (int ci = 0; ci < Cin; ci++) {
    const float* ip = in + ((size_t)b * Cin + ci) * Lin + p;
    const float* wp = w + ((size_t)co * Cin + ci) * 8;
    #pragma unroll
    for (int k = 0; k < 8; k++) acc += ip[k] * wp[k];
  }
  out[i] = fmaxf(acc, 0.0f);
}
__global__ void k_pool3(const float* __restrict__ in, float* __restrict__ out,
                        int Bn, int C, int Lin, int Lp) {
  int i = blockIdx.x * blockDim.x + threadIdx.x;
  if (i >= Bn * C * Lp) return;
  int p = i % Lp; int bc = i / Lp;
  const float* ip = in + (size_t)bc * Lin + p * 3;
  out[i] = fmaxf(ip[0], fmaxf(ip[1], ip[2]));
}

// ------------------------------ small MHA ----------------------------------
__global__ void k_tge(const float* __restrict__ gec, float* __restrict__ qseq) {
  int i = blockIdx.x * blockDim.x + threadIdx.x;
  if (i >= 256 * 33 * 128) return;
  int c = i & 127, l = (i >> 7) % 33, b = i / (33 * 128);
  qseq[i] = gec[((size_t)b * 128 + c) * 33 + l];
}
__global__ void k_kv(const float* __restrict__ methc, const float* __restrict__ mutc,
                     float* __restrict__ kvseq) {
  int i = blockIdx.x * blockDim.x + threadIdx.x;
  if (i >= 256 * 33 * 128) return;
  int c = i & 127, l = (i >> 7) % 33, b = i / (33 * 128);
  kvseq[i] = (l < 10) ? methc[((size_t)b * 128 + c) * 10 + l]
                      : mutc[((size_t)b * 128 + c) * 23 + (l - 10)];
}
__global__ void k_mha(const float* __restrict__ Qp, const float* __restrict__ Kp,
                      const float* __restrict__ Vp, float* __restrict__ oseq) {
  int i = blockIdx.x * blockDim.x + threadIdx.x;
  if (i >= 256 * 8 * 33) return;
  int lq = i % 33, h = (i / 33) % 8, b = i / (33 * 8);
  const float* Qr = Qp + ((size_t)b * 33 + lq) * 128 + h * 16;
  float s[33], mx = -1e30f;
  for (int k = 0; k < 33; k++) {
    const float* Kr = Kp + ((size_t)b * 33 + k) * 128 + h * 16;
    float acc = 0.0f;
    #pragma unroll
    for (int dd = 0; dd < 16; dd++) acc += Qr[dd] * Kr[dd];
    s[k] = acc * 0.25f;
    mx = fmaxf(mx, s[k]);
  }
  float sum = 0.0f;
  for (int k = 0; k < 33; k++) { s[k] = expf(s[k] - mx); sum += s[k]; }
  float inv = 1.0f / sum;
  for (int dd = 0; dd < 16; dd++) {
    float o = 0.0f;
    for (int k = 0; k < 33; k++)
      o += s[k] * Vp[((size_t)b * 33 + k) * 128 + h * 16 + dd];
    oseq[((size_t)b * 33 + lq) * 128 + h * 16 + dd] = o * inv;
  }
}

// -------------------------------- head -------------------------------------
__global__ void k_concat4(const float* a, const float* b, const float* c,
                          const float* d, float* out) {
  int i = blockIdx.x * blockDim.x + threadIdx.x;
  if (i >= 256 * 512) return;
  int r = i >> 9, col = i & 511;
  float v;
  if (col < 128)      v = a[r * 128 + col];
  else if (col < 256) v = b[r * 128 + col - 128];
  else if (col < 384) v = c[r * 128 + col - 256];
  else                v = d[r * 128 + col - 384];
  out[i] = v;
}
__global__ void k_outhead(const float* __restrict__ x, const float* __restrict__ w,
                          const float* __restrict__ b, float* __restrict__ o) {
  int t = blockIdx.x * blockDim.x + threadIdx.x;
  if (t >= 256) return;
  float acc = b[0];
  for (int j = 0; j < 128; j++) acc += x[t * 128 + j] * w[j];
  o[t] = acc;
}

// ===========================================================================
// Host side
// ===========================================================================
static inline int cdiv(int a, int b) { return (a + b - 1) / b; }
static inline int pad64h(int x)  { return cdiv(x, 64) * 64; }

// params leaves in jax pytree (alphabetical DFS) order
static const int kLeafSz[80] = {
  1024, 524288, 128, 131072, 1500, 2340000, 128, 192000,
  780, 780, 780, 60840, 780, 608400,
  32, 256, 64, 16384, 128, 65536, 128, 540672,
  32, 256, 64, 16384, 128, 65536, 128, 163840,
  128, 128, 128, 128, 16384, 16384, 16384, 16384,
  32, 256, 64, 16384, 128, 65536, 128, 376832,
  1, 128,
  78, 78, 78, 78, 2048, 78, 159744, 159744, 78, 78, 78, 78, 6084, 6084, 6084, 6084,
  780, 780, 780, 780, 2048, 780, 1597440, 1597440, 780, 780, 780, 780,
  608400, 608400, 608400, 608400
};

extern "C" void kernel_launch(void* const* d_in, const int* in_sizes, int n_in,
                              void* d_out, int out_size, void* d_ws, size_t ws_size,
                              hipStream_t stream) {
  (void)in_sizes; (void)out_size; (void)ws_size;

  const float* X      = (const float*)d_in[0];
  const float* T_MUT  = (const float*)d_in[1];
  const float* T_METH = (const float*)d_in[2];
  const float* T_GE   = (const float*)d_in[3];
  const int*   EI     = (const int*)d_in[n_in - 2];
  const int*   BATCH  = (const int*)d_in[n_in - 1];

  const float* leaf[80];
  if (n_in >= 4 + 80 + 2) {
    for (int i = 0; i < 80; i++) leaf[i] = (const float*)d_in[4 + i];
  } else {
    const float* base = (const float*)d_in[4];
    long long off = 0;
    for (int i = 0; i < 80; i++) { leaf[i] = base + off; off += kLeafSz[i]; }
  }

  // ------------------------- workspace arena --------------------------------
  char* wsb = (char*)d_ws; size_t off = 0;
  auto alloc = [&](size_t n) -> char* {
    size_t a = (off + 255) & ~(size_t)255; off = a + n; return wsb + a;
  };
  _Float16* H16   = (_Float16*)alloc((size_t)MP * 832 * 2);
  _Float16* Q16   = (_Float16*)alloc((size_t)MP * 832 * 2);
  _Float16* K16   = (_Float16*)alloc((size_t)MP * 832 * 2);
  _Float16* V16   = (_Float16*)alloc((size_t)MP * 832 * 2);   // holds V^T
  _Float16* F16   = (_Float16*)alloc((size_t)MP * FFH * 2);
  _Float16* W16   = (_Float16*)alloc((size_t)2457600 * 2);    // max padded weight
  _Float16* Seq16 = (_Float16*)alloc((size_t)8448 * 128 * 2);
  _Float16* Sm16  = (_Float16*)alloc((size_t)256 * 4224 * 2);
  float* Bq   = (float*)alloc((size_t)NN * D2 * 4);
  float* Bk   = (float*)alloc((size_t)NN * D2 * 4);
  float* Bv   = (float*)alloc((size_t)NN * D2 * 4);
  float* Batt = (float*)alloc((size_t)NN * D2 * 4);
  float* Bo   = (float*)alloc((size_t)NN * D2 * 4);
  float* Btmp = (float*)alloc((size_t)NN * D2 * 4);
  float* Bffn = (float*)alloc((size_t)NN * FFH * 4);
  float* Hb1  = (float*)alloc((size_t)NN * D2 * 4);
  float* Hb2  = (float*)alloc((size_t)NN * D2 * 4);
  float*    AS    = (float*)alloc((size_t)NN * 10 * 4);
  float*    AD    = (float*)alloc((size_t)NN * 10 * 4);
  unsigned* MKEY  = (unsigned*)alloc((size_t)NN * 10 * 4);
  float*    DEN   = (float*)alloc((size_t)NN * 10 * 4);
  float*    EXS   = (float*)alloc((size_t)NEDGE * 10 * 4);
  float*    DEG   = (float*)alloc((size_t)NN * 4);
  unsigned* GMAXK = (unsigned*)alloc((size_t)256 * D2 * 4);
  float*    GSUM  = (float*)alloc((size_t)256 * D2 * 4);
  float*    CNT   = (float*)alloc((size_t)256 * 4);
  float*    XG    = (float*)alloc((size_t)256 * 1560 * 4);
  float*    XG1   = (float*)alloc((size_t)256 * 1500 * 4);
  float*    XG2   = (float*)alloc((size_t)256 * 128 * 4);
  float*    XTMUT = (float*)alloc((size_t)256 * 128 * 4);
  float*    XTMET = (float*)alloc((size_t)256 * 128 * 4);
  float*    XTGE  = (float*)alloc((size_t)256 * 128 * 4);
  float*    XC    = (float*)alloc((size_t)256 * 512 * 4);
  float*    FC1O  = (float*)alloc((size_t)256 * 1024 * 4);
  float*    FC2O  = (float*)alloc((size_t)256 * 128 * 4);
  float*    MUTC  = (float*)alloc((size_t)256 * 128 * 23 * 4);
  float*    METHC = (float*)alloc((size_t)256 * 128 * 10 * 4);
  float*    GEC   = (float*)alloc((size_t)256 * 128 * 33 * 4);
  // large f32 buffers reused after the graph stage (stream-serial):
  float* convA = Bffn;  float* convB = Bq;
  float* QSEQ = Bk;     float* KVSEQ = Bv;
  float* QP = Batt;     float* KP = Bo;   float* VP = Btmp;
  float* OSEQ = Hb1;    float* OM = Hb2;

  // ----------------------------- helpers ------------------------------------
  auto cvt = [&](const float* s, _Float16* d, int R, int C, int Rp, int Cp) {
    int total = Rp * Cp;
    k_cvt_pad<<<cdiv(total, 256), 256, 0, stream>>>(s, d, R, C, Rp, Cp);
  };
  auto cvtT = [&](const float* s, _Float16* d, int R, int C, int Rp, int Cp) {
    int total = Rp * Cp;
    k_cvt_padT<<<cdiv(total, 256), 256, 0, stream>>>(s, d, R, C, Rp, Cp);
  };
  auto gemm = [&](const _Float16* A16, int M, int Kt, const float* W, int Nt,
                  float* Dst, const float* bias, bool relu) {
    int Kp = pad64h(Kt), Np = pad64h(Nt);
    cvtT(W, W16, Kt, Nt, Kp, Np);                 // W16 = B^T : [Np][Kp]
    dim3 g(cdiv(Nt, 64), cdiv(M, 128));
    k_gemm<<<g, 256, 0, stream>>>(A16, W16, Dst, bias, M, Nt, Kp, Kp, Kp,
                                  relu ? 1 : 0);
  };
  auto fillf = [&](float* p, float v, int n) {
    k_fillf<<<cdiv(n, 256), 256, 0, stream>>>(p, v, n);
  };
  auto fillu = [&](unsigned* p, unsigned v, int n) {
    k_fillu<<<cdiv(n, 256), 256, 0, stream>>>(p, v, n);
  };

  // transformer layer (hin -> hout), L = 16 tl leaves alphabetical:
  // 0 bk,1 bo,2 bq,3 bv,4 f_b1,5 f_b2,6 f_w1,7 f_w2,
  // 8 ln1_b,9 ln1_g,10 ln2_b,11 ln2_g,12 wk,13 wo,14 wq,15 wv
  auto run_tl = [&](const float* hin, float* hout, int d, const float* const* L) {
    int dp = pad64h(d);
    cvt(hin, H16, NN, d, MP, dp);
    gemm(H16, NN, d, L[14], d, Bq, L[2], false);
    gemm(H16, NN, d, L[12], d, Bk, L[0], false);
    gemm(H16, NN, d, L[15], d, Bv, L[3], false);
    cvt(Bq, Q16, NN, d, MP, dp);
    cvt(Bk, K16, NN, d, MP, dp);
    cvtT(Bv, V16, NN, d, MP, dp);                  // V16 = V^T : [dp][MP]
    size_t shm = (size_t)16 * dp * 4 + 16 * 32 * 2 + 16 * 4;
    k_attn<<<NN / 16, 32, shm, stream>>>(Q16, K16, V16, Batt, NN, NN, MP, d, dp,
                                         1.0f / sqrtf((float)d));
    cvt(Batt, Q16, NN, d, MP, dp);
    gemm(Q16, NN, d, L[13], d, Bo, L[1], false);
    k_resln<<<NN, 256, 0, stream>>>(hin, Bo, L[9], L[8], Btmp, d);
    cvt(Btmp, H16, NN, d, MP, dp);
    gemm(H16, NN, d, L[6], FFH, Bffn, L[4], true);
    cvt(Bffn, F16, NN, FFH, MP, FFH);
    gemm(F16, NN, FFH, L[7], d, Bo, L[5], false);
    k_resln<<<NN, 256, 0, stream>>>(Btmp, Bo, L[11], L[10], hout, d);
  };

  auto run_branch = [&](const float* tin, int L, const float* const* BP,
                        float* outc) {
    int L1 = L - 7,  P1 = L1 / 3;
    int L2 = P1 - 7, P2 = L2 / 3;
    int L3 = P2 - 7, P3 = L3 / 3;
    k_conv1d<<<cdiv(256 * 32 * L1, 256), 256, 0, stream>>>(tin, BP[1], BP[0],
        convA, 256, 1, L, 32, L1);
    k_pool3<<<cdiv(256 * 32 * P1, 256), 256, 0, stream>>>(convA, convB, 256, 32, L1, P1);
    k_conv1d<<<cdiv(256 * 64 * L2, 256), 256, 0, stream>>>(convB, BP[3], BP[2],
        convA, 256, 32, P1, 64, L2);
    k_pool3<<<cdiv(256 * 64 * P2, 256), 256, 0, stream>>>(convA, convB, 256, 64, L2, P2);
    k_conv1d<<<cdiv(256 * 128 * L3, 256), 256, 0, stream>>>(convB, BP[5], BP[4],
        convA, 256, 64, P2, 128, L3);
    k_pool3<<<cdiv(256 * 128 * P3, 256), 256, 0, stream>>>(convA, outc, 256, 128, L3, P3);
  };

  // ------------------------------ pipeline ----------------------------------
  // 1) transformer layer 1 (d=78): X -> Hb1
  const float* TL1[16]; for (int i = 0; i < 16; i++) TL1[i] = leaf[48 + i];
  run_tl(X, Hb1, D1, TL1);

  // 2) GAT: Hb1 -> Hb2
  cvt(Hb1, H16, NN, D1, MP, 128);
  gemm(H16, NN, D1, leaf[11] /*gat.w*/, D2, Bq /*hw*/, nullptr, false);
  k_gatcoef<<<cdiv(NN * 10, 256), 256, 0, stream>>>(Bq, leaf[9], leaf[8], AS, AD, NN);
  fillu(MKEY, 0u, NN * 10);
  k_gatmax<<<cdiv(NEDGE * 10, 256), 256, 0, stream>>>(EI, AS, AD, MKEY);
  fillf(DEN, 0.0f, NN * 10);
  k_gatexp<<<cdiv(NEDGE * 10, 256), 256, 0, stream>>>(EI, AS, AD, MKEY, EXS, DEN);
  fillf(Hb2, 0.0f, NN * D2);
  k_gatscat<<<cdiv(NEDGE * D2, 256), 256, 0, stream>>>(EI, EXS, DEN, Bq, Hb2);
  k_biasrelu<<<cdiv(NN * D2, 256), 256, 0, stream>>>(Hb2, leaf[10], NN, D2);

  // 3) transformer layer 2 (d=780): Hb2 -> Hb1
  const float* TL2[16]; for (int i = 0; i < 16; i++) TL2[i] = leaf[64 + i];
  run_tl(Hb2, Hb1, D2, TL2);

  // 4) GCN: Hb1 -> Hb2
  fillf(DEG, 0.0f, NN);
  k_deg<<<cdiv(NEDGE, 256), 256, 0, stream>>>(EI, DEG);
  cvt(Hb1, H16, NN, D2, MP, 832);
  gemm(H16, NN, D2, leaf[13] /*gcn.w*/, D2, Bq, nullptr, false);
  fillf(Hb2, 0.0f, NN * D2);
  k_gcnscat<<<cdiv(NEDGE * D2, 256), 256, 0, stream>>>(EI, DEG, Bq, Hb2);
  k_biasrelu<<<cdiv(NN * D2, 256), 256, 0, stream>>>(Hb2, leaf[12], NN, D2);

  // 5) pooling + graph FCs -> XG2
  fillu(GMAXK, 0u, 256 * D2); fillf(GSUM, 0.0f, 256 * D2); fillf(CNT, 0.0f, 256);
  k_cnt<<<cdiv(NN, 256), 256, 0, stream>>>(BATCH, CNT, NN);
  k_pool<<<cdiv(NN * D2, 256), 256, 0, stream>>>(Hb2, BATCH, GMAXK, GSUM, NN, D2);
  k_xg<<<cdiv(256 * 1560, 256), 256, 0, stream>>>(GMAXK, GSUM, CNT, XG, D2);
  cvt(XG, Sm16, 256, 1560, 256, 1600);
  gemm(Sm16, 256, 1560, leaf[5], 1500, XG1, leaf[4], true);
  cvt(XG1, Sm16, 256, 1500, 256, 1536);
  gemm(Sm16, 256, 1500, leaf[7], 128, XG2, leaf[6], false);

  // 6) conv branches (clobber Bffn/Bq, already consumed)
  const float* MUTP[8];  for (int i = 0; i < 8; i++) MUTP[i]  = leaf[38 + i];
  const float* METHP[8]; for (int i = 0; i < 8; i++) METHP[i] = leaf[22 + i];
  const float* GEP[8];   for (int i = 0; i < 8; i++) GEP[i]   = leaf[14 + i];
  run_branch(T_MUT,  735,  MUTP,  MUTC);
  run_branch(T_METH, 377,  METHP, METHC);
  run_branch(T_GE,   1000, GEP,   GEC);
  cvt(MUTC, Sm16, 256, 2944, 256, 2944);
  gemm(Sm16, 256, 2944, MUTP[7], 128, XTMUT, MUTP[6], false);
  cvt(METHC, Sm16, 256, 1280, 256, 1280);
  gemm(Sm16, 256, 1280, METHP[7], 128, XTMET, METHP[6], false);

  // 7) MHA over conv features -> XTGE
  k_tge<<<cdiv(256 * 33 * 128, 256), 256, 0, stream>>>(GEC, QSEQ);
  k_kv<<<cdiv(256 * 33 * 128, 256), 256, 0, stream>>>(METHC, MUTC, KVSEQ);
  cvt(QSEQ, Seq16, 8448, 128, 8448, 128);
  gemm(Seq16, 8448, 128, leaf[36], 128, QP, leaf[32], false);   // wq,bq
  cvt(KVSEQ, Seq16, 8448, 128, 8448, 128);
  gemm(Seq16, 8448, 128, leaf[34], 128, KP, leaf[30], false);   // wk,bk
  gemm(Seq16, 8448, 128, leaf[37], 128, VP, leaf[33], false);   // wv,bv
  k_mha<<<cdiv(256 * 8 * 33, 256), 256, 0, stream>>>(QP, KP, VP, OSEQ);
  cvt(OSEQ, Seq16, 8448, 128, 8448, 128);
  gemm(Seq16, 8448, 128, leaf[35], 128, OM, leaf[31], false);   // wo,bo
  cvt(OM, Sm16, 256, 4224, 256, 4224);
  gemm(Sm16, 256, 4224, GEP[7], 128, XTGE, GEP[6], false);

  // 8) head -> d_out (256x1)
  k_concat4<<<cdiv(256 * 512, 256), 256, 0, stream>>>(XG2, XTMUT, XTMET, XTGE, XC);
  cvt(XC, Sm16, 256, 512, 256, 512);
  gemm(Sm16, 256, 512, leaf[1], 1024, FC1O, leaf[0], true);     // fc1
  cvt(FC1O, Sm16, 256, 1024, 256, 1024);
  gemm(Sm16, 256, 1024, leaf[3], 128, FC2O, leaf[2], true);     // fc2
  k_outhead<<<1, 256, 0, stream>>>(FC2O, leaf[47], leaf[46], (float*)d_out);
}
